// MMFuser_85083302134338
// MI455X (gfx1250) — matrix operand order repaired and checked by reading, not compiled, not run on hardware
//
#include <hip/hip_runtime.h>

#define EMBED 1024
#define BATCH 8
#define SQLEN 577
#define SKLEN 13271
#define SKPAD 13280   // 415 * 32
#define SQPAD 608     // 19 * 32

typedef __attribute__((ext_vector_type(16))) __bf16 bf16x16;
typedef __attribute__((ext_vector_type(8)))  float  f32x8;

static __device__ __forceinline__ unsigned short f2bf(float x) {
  unsigned int u = __builtin_bit_cast(unsigned int, x);
  u += 0x7FFFu + ((u >> 16) & 1u);   // round-to-nearest-even
  return (unsigned short)(u >> 16);
}

static __device__ __forceinline__ bf16x16 pack_ab(uint4 lo, uint4 hi) {
  union { uint4 u[2]; bf16x16 v; } c;
  c.u[0] = lo; c.u[1] = hi;
  return c.v;
}

static __device__ __forceinline__ f32x8 wmma_bf16(bf16x16 a, bf16x16 b, f32x8 c) {
  return __builtin_amdgcn_wmma_f32_16x16x32_bf16(false, a, false, b, (short)0, c,
                                                 false, false);
}

static __device__ __forceinline__ int imin(int a, int b) { return a < b ? a : b; }

// ---------------------------------------------------------------------------
// LayerNorm over last dim (1024), f32 in -> bf16 out (+ optional f32 out).
// One block (256 threads) per row; each thread owns 4 contiguous elements.
// ---------------------------------------------------------------------------
__global__ __launch_bounds__(256) void ln_kernel(
    const float* __restrict__ x, const float* __restrict__ w,
    const float* __restrict__ bias, unsigned short* __restrict__ out_bf,
    float* __restrict__ out_f32)
{
  const int row  = blockIdx.x;
  const int t    = threadIdx.x;
  const int wave = t >> 5, lane = t & 31;
  __shared__ float red[16];

  const float4 v = ((const float4*)(x + (size_t)row * EMBED))[t];
  float s = v.x + v.y + v.z + v.w;
  s += __shfl_xor(s, 1);  s += __shfl_xor(s, 2);  s += __shfl_xor(s, 4);
  s += __shfl_xor(s, 8);  s += __shfl_xor(s, 16);
  if (lane == 0) red[wave] = s;
  __syncthreads();
  float tot = 0.f;
#pragma unroll
  for (int i = 0; i < 8; ++i) tot += red[i];
  const float mean = tot * (1.0f / EMBED);

  float4 c;
  c.x = v.x - mean; c.y = v.y - mean; c.z = v.z - mean; c.w = v.w - mean;
  float sq = c.x * c.x + c.y * c.y + c.z * c.z + c.w * c.w;
  sq += __shfl_xor(sq, 1);  sq += __shfl_xor(sq, 2);  sq += __shfl_xor(sq, 4);
  sq += __shfl_xor(sq, 8);  sq += __shfl_xor(sq, 16);
  if (lane == 0) red[8 + wave] = sq;
  __syncthreads();
  float tot2 = 0.f;
#pragma unroll
  for (int i = 0; i < 8; ++i) tot2 += red[8 + i];
  const float rstd = rsqrtf(tot2 * (1.0f / EMBED) + 1e-5f);

  const float4 wv = ((const float4*)w)[t];
  const float4 bv = ((const float4*)bias)[t];
  float4 y;
  y.x = c.x * rstd * wv.x + bv.x;
  y.y = c.y * rstd * wv.y + bv.y;
  y.z = c.z * rstd * wv.z + bv.z;
  y.w = c.w * rstd * wv.w + bv.w;

  uint2 pk;
  pk.x = (unsigned)f2bf(y.x) | ((unsigned)f2bf(y.y) << 16);
  pk.y = (unsigned)f2bf(y.z) | ((unsigned)f2bf(y.w) << 16);
  ((uint2*)out_bf)[(size_t)row * 256 + t] = pk;
  if (out_f32) ((float4*)out_f32)[(size_t)row * 256 + t] = y;
}

// ---------------------------------------------------------------------------
// bf16 transpose: in [B, R, 1024] -> out [B, 1024, Rpad]; pad rows zeroed.
// 32x32 tiles, block (32x8), grid (1024/32, Rpad/32, B).
// ---------------------------------------------------------------------------
__global__ __launch_bounds__(256) void transpose_bf16_kernel(
    const unsigned short* __restrict__ in, unsigned short* __restrict__ out,
    int R, int Rpad)
{
  __shared__ unsigned short tile[32][33];
  const int b  = blockIdx.z;
  const int c0 = blockIdx.x * 32;
  const int r0 = blockIdx.y * 32;
  const int tx = threadIdx.x & 31;
  const int ty = threadIdx.x >> 5;
#pragma unroll
  for (int i = 0; i < 4; ++i) {
    const int r = r0 + ty + i * 8;
    tile[ty + i * 8][tx] =
        (r < R) ? in[((size_t)b * R + r) * EMBED + c0 + tx] : (unsigned short)0;
  }
  __syncthreads();
#pragma unroll
  for (int i = 0; i < 4; ++i) {
    const int c = c0 + ty + i * 8;
    out[((size_t)b * EMBED + c) * Rpad + r0 + tx] = tile[tx][ty + i * 8];
  }
}

// ---------------------------------------------------------------------------
// Flash attention, bf16 WMMA, f32 accumulate. out = softmax(Q K^T / 32) V.
// Block = 128 threads (4 waves), one 16-row q-tile per block.
// Q slice held in registers (loop-invariant A operands). QK^T: d=1024 split
// across 4 waves, partial 16x32 score tiles reduced through LDS. Softmax
// redundant per wave. PV: each wave owns a 256-wide d-slice; B operands read
// contiguously from the transposed KV copy. B loads double-buffered so each
// WMMA runs behind prefetched loads.
// ---------------------------------------------------------------------------
__global__ __launch_bounds__(128) void flash_attn_kernel(
    const unsigned short* __restrict__ Q,    // [B, SQ, 1024] bf16
    const unsigned short* __restrict__ K,    // [B, SK, 1024] bf16 (K == V)
    const unsigned short* __restrict__ KT,   // [B, 1024, SKpad] bf16
    float* __restrict__ O,                   // [B, SQ, 1024] f32
    int SQ, int SK, int SKpad)
{
  __shared__ float4 Ex[4][2][32][2];                     // 8 KB score exchange
  __shared__ alignas(16) unsigned short Ps[16][32];      // 1 KB P (bf16)

  const int b     = blockIdx.y;
  const int q0    = blockIdx.x * 16;
  const int tid   = threadIdx.x;
  const int wave  = tid >> 5;
  const int lane  = tid & 31;
  const int half  = lane >> 4;
  const int lm    = lane & 15;
  const int dbase = wave * 256;

  // ---- loop-invariant Q A-operands for this wave's 256-wide d slice ----
  bf16x16 qa[8];
  {
    const int qr = imin(q0 + lm, SQ - 1);
    const unsigned short* qrow = Q + ((size_t)b * SQ + qr) * EMBED + dbase;
#pragma unroll
    for (int st = 0; st < 8; ++st) {
      qa[st] = pack_ab(*(const uint4*)(qrow + st * 32 + half * 16),
                       *(const uint4*)(qrow + st * 32 + half * 16 + 8));
    }
  }

  f32x8 zero;
#pragma unroll
  for (int r = 0; r < 8; ++r) zero[r] = 0.0f;
  f32x8 acc[16];
#pragma unroll
  for (int t = 0; t < 16; ++t) acc[t] = zero;
  float mrow[8], lrow[8];
#pragma unroll
  for (int r = 0; r < 8; ++r) { mrow[r] = -3.0e38f; lrow[r] = 0.0f; }

  const unsigned short* Kb  = K  + (size_t)b * SK * EMBED;
  const unsigned short* KTb = KT + (size_t)b * EMBED * SKpad;
  const size_t vstride = (size_t)16 * SKpad;   // elements between PV N-tiles
  const int nkt = (SK + 31) >> 5;

  for (int kt = 0; kt < nkt; ++kt) {
    const int k0 = kt << 5;

    // ---- QK^T partial (this wave's 256-wide d slice), 16x32 scores ----
    const unsigned short* Kr0 =
        Kb + (size_t)imin(k0 + lm, SK - 1) * EMBED + dbase + half * 16;
    const unsigned short* Kr1 =
        Kb + (size_t)imin(k0 + 16 + lm, SK - 1) * EMBED + dbase + half * 16;

    f32x8 s0 = zero, s1 = zero;
    uint4 p0a = *(const uint4*)(Kr0);
    uint4 p0b = *(const uint4*)(Kr0 + 8);
    uint4 p1a = *(const uint4*)(Kr1);
    uint4 p1b = *(const uint4*)(Kr1 + 8);
#pragma unroll
    for (int st = 0; st < 8; ++st) {
      const uint4 c0a = p0a, c0b = p0b, c1a = p1a, c1b = p1b;
      if (st < 7) {   // prefetch next d-step for both score columns
        const int e = (st + 1) * 32;
        p0a = *(const uint4*)(Kr0 + e);
        p0b = *(const uint4*)(Kr0 + e + 8);
        p1a = *(const uint4*)(Kr1 + e);
        p1b = *(const uint4*)(Kr1 + e + 8);
      }
      s0 = wmma_bf16(qa[st], pack_ab(c0a, c0b), s0);
      s1 = wmma_bf16(qa[st], pack_ab(c1a, c1b), s1);
    }
    {
      union { f32x8 v; float4 f[2]; } c0, c1;
      c0.v = s0; c1.v = s1;
      Ex[wave][0][lane][0] = c0.f[0]; Ex[wave][0][lane][1] = c0.f[1];
      Ex[wave][1][lane][0] = c1.f[0]; Ex[wave][1][lane][1] = c1.f[1];
    }
    __syncthreads();

    // ---- reduce partials over 4 waves (each wave gets full scores) ----
    float fs0[8], fs1[8];
#pragma unroll
    for (int r = 0; r < 8; ++r) { fs0[r] = 0.f; fs1[r] = 0.f; }
#pragma unroll
    for (int wv = 0; wv < 4; ++wv) {
      const float4 a0 = Ex[wv][0][lane][0], a1 = Ex[wv][0][lane][1];
      const float4 b0 = Ex[wv][1][lane][0], b1 = Ex[wv][1][lane][1];
      fs0[0] += a0.x; fs0[1] += a0.y; fs0[2] += a0.z; fs0[3] += a0.w;
      fs0[4] += a1.x; fs0[5] += a1.y; fs0[6] += a1.z; fs0[7] += a1.w;
      fs1[0] += b0.x; fs1[1] += b0.y; fs1[2] += b0.z; fs1[3] += b0.w;
      fs1[4] += b1.x; fs1[5] += b1.y; fs1[6] += b1.z; fs1[7] += b1.w;
    }
    const bool msk0 = (k0 + lm)      >= SK;
    const bool msk1 = (k0 + 16 + lm) >= SK;
#pragma unroll
    for (int r = 0; r < 8; ++r) {
      fs0[r] = msk0 ? -3.0e38f : fs0[r] * 0.03125f;   // scale = D^-0.5 = 1/32
      fs1[r] = msk1 ? -3.0e38f : fs1[r] * 0.03125f;
    }

    // ---- online softmax (row = vgpr + 8*half; reduce within 16-lane half) --
    float alpha[8];
#pragma unroll
    for (int r = 0; r < 8; ++r) {
      float cm = fmaxf(fs0[r], fs1[r]);
      cm = fmaxf(cm, __shfl_xor(cm, 1));
      cm = fmaxf(cm, __shfl_xor(cm, 2));
      cm = fmaxf(cm, __shfl_xor(cm, 4));
      cm = fmaxf(cm, __shfl_xor(cm, 8));
      const float nm = fmaxf(mrow[r], cm);
      alpha[r] = __expf(mrow[r] - nm);
      mrow[r] = nm;
      fs0[r] = __expf(fs0[r] - nm);
      fs1[r] = __expf(fs1[r] - nm);
      float rs = fs0[r] + fs1[r];
      rs += __shfl_xor(rs, 1);
      rs += __shfl_xor(rs, 2);
      rs += __shfl_xor(rs, 4);
      rs += __shfl_xor(rs, 8);
      lrow[r] = lrow[r] * alpha[r] + rs;
    }
#pragma unroll
    for (int t = 0; t < 16; ++t)
#pragma unroll
      for (int r = 0; r < 8; ++r) acc[t][r] *= alpha[r];

    // ---- repack P to bf16 row-major in LDS (wave 0; all waves identical) ---
    if (wave == 0) {
#pragma unroll
      for (int r = 0; r < 8; ++r) {
        Ps[r + 8 * half][lm]      = f2bf(fs0[r]);
        Ps[r + 8 * half][16 + lm] = f2bf(fs1[r]);
      }
    }
    __syncthreads();

    // ---- PV: this wave's 256-wide d slice, 16 N-tiles, K=32 ----
    const bf16x16 ap = pack_ab(*(const uint4*)(&Ps[lm][half * 16]),
                               *(const uint4*)(&Ps[lm][half * 16 + 8]));
    const unsigned short* Vp =
        KTb + (size_t)(dbase + lm) * SKpad + k0 + half * 16;
    uint4 va = *(const uint4*)(Vp);
    uint4 vb = *(const uint4*)(Vp + 8);
#pragma unroll
    for (int t = 0; t < 16; ++t) {
      const uint4 ca = va, cb = vb;
      if (t < 15) {   // prefetch next d N-tile
        const unsigned short* nxt = Vp + (size_t)(t + 1) * vstride;
        va = *(const uint4*)(nxt);
        vb = *(const uint4*)(nxt + 8);
      }
      acc[t] = wmma_bf16(ap, pack_ab(ca, cb), acc[t]);
    }
  }

  float inv[8];
#pragma unroll
  for (int r = 0; r < 8; ++r) inv[r] = 1.0f / lrow[r];
#pragma unroll
  for (int t = 0; t < 16; ++t) {
#pragma unroll
    for (int r = 0; r < 8; ++r) {
      const int q = q0 + r + 8 * half;
      if (q < SQ)
        O[((size_t)b * SQ + q) * EMBED + dbase + t * 16 + lm] = acc[t][r] * inv[r];
    }
  }
}

// ---------------------------------------------------------------------------
// MLP: h = gelu_erf(x @ fc1_w^T + fc1_b) (H=4); out = h @ fc2_w^T + fc2_b.
// One block (256 threads) per row.
// ---------------------------------------------------------------------------
__global__ __launch_bounds__(256) void mlp_kernel(
    const float* __restrict__ att, const float* __restrict__ fc1w,
    const float* __restrict__ fc1b, const float* __restrict__ fc2w,
    const float* __restrict__ fc2b, float* __restrict__ out)
{
  const int row  = blockIdx.x;
  const int t    = threadIdx.x;
  const int wave = t >> 5, lane = t & 31;
  __shared__ float red[8][4];
  __shared__ float hsh[4];

  const float* xr = att + (size_t)row * EMBED;
  float a[4] = {0.f, 0.f, 0.f, 0.f};
#pragma unroll
  for (int i = 0; i < 4; ++i) {
    const int d = t + i * 256;
    const float xv = xr[d];
#pragma unroll
    for (int j = 0; j < 4; ++j) a[j] += xv * fc1w[j * EMBED + d];
  }
#pragma unroll
  for (int j = 0; j < 4; ++j) {
    a[j] += __shfl_xor(a[j], 1);
    a[j] += __shfl_xor(a[j], 2);
    a[j] += __shfl_xor(a[j], 4);
    a[j] += __shfl_xor(a[j], 8);
    a[j] += __shfl_xor(a[j], 16);
  }
  if (lane == 0) {
#pragma unroll
    for (int j = 0; j < 4; ++j) red[wave][j] = a[j];
  }
  __syncthreads();
  if (t < 4) {
    float z = fc1b[t];
#pragma unroll
    for (int wv = 0; wv < 8; ++wv) z += red[wv][t];
    hsh[t] = 0.5f * z * (1.0f + erff(z * 0.70710678118654752f));
  }
  __syncthreads();
  const float h0 = hsh[0], h1 = hsh[1], h2 = hsh[2], h3 = hsh[3];
#pragma unroll
  for (int i = 0; i < 4; ++i) {
    const int d = t + i * 256;
    out[(size_t)row * EMBED + d] = fc2b[d] + h0 * fc2w[d * 4 + 0] +
                                   h1 * fc2w[d * 4 + 1] + h2 * fc2w[d * 4 + 2] +
                                   h3 * fc2w[d * 4 + 3];
  }
}

// out = norm_F_L + gamma2 * (cross_attended + gamma1 * self_attended)
__global__ __launch_bounds__(256) void final_kernel(
    const float* __restrict__ nfl, const float* __restrict__ ca,
    const float* __restrict__ sa, const float* __restrict__ g1,
    const float* __restrict__ g2, float* __restrict__ out, long long n)
{
  const long long i = (long long)blockIdx.x * 256 + threadIdx.x;
  if (i >= n) return;
  const int d = (int)(i & (EMBED - 1));
  out[i] = nfl[i] + g2[d] * (ca[i] + g1[d] * sa[i]);
}

extern "C" void kernel_launch(void* const* d_in, const int* in_sizes, int n_in,
                              void* d_out, int out_size, void* d_ws, size_t ws_size,
                              hipStream_t stream)
{
  (void)in_sizes; (void)n_in; (void)out_size; (void)ws_size;
  const float* input_features = (const float*)d_in[0];
  const float* query_feature  = (const float*)d_in[1];
  const float* norm1_w = (const float*)d_in[2];
  const float* norm1_b = (const float*)d_in[3];
  const float* norm2_w = (const float*)d_in[4];
  const float* norm2_b = (const float*)d_in[5];
  const float* norm3_w = (const float*)d_in[6];
  const float* norm3_b = (const float*)d_in[7];
  const float* gamma1  = (const float*)d_in[8];
  const float* gamma2  = (const float*)d_in[9];
  const float* fc1_w   = (const float*)d_in[10];
  const float* fc1_b   = (const float*)d_in[11];
  const float* fc2_w   = (const float*)d_in[12];
  const float* fc2_b   = (const float*)d_in[13];
  float* out = (float*)d_out;

  char* base = (char*)d_ws;
  size_t off = 0;
  auto alloc = [&](size_t bytes) -> char* {
    char* p = base + off;
    off = (off + bytes + 255) & ~((size_t)255);
    return p;
  };
  const size_t nQ = (size_t)BATCH * SQLEN * EMBED;
  const size_t nK = (size_t)BATCH * SKLEN * EMBED;
  unsigned short* normX   = (unsigned short*)alloc(nK * 2);
  unsigned short* normXT  = (unsigned short*)alloc((size_t)BATCH * EMBED * SKPAD * 2);
  unsigned short* normFLb = (unsigned short*)alloc(nQ * 2);
  float*          normFLf = (float*)alloc(nQ * 4);
  float*          att     = (float*)alloc(nQ * 4);
  float*          cross   = (float*)alloc(nQ * 4);
  unsigned short* ncrossb = (unsigned short*)alloc(nQ * 2);
  unsigned short* ncrossT = (unsigned short*)alloc((size_t)BATCH * EMBED * SQPAD * 2);
  float*          selfatt = (float*)alloc(nQ * 4);

  ln_kernel<<<dim3(BATCH * SKLEN), dim3(256), 0, stream>>>(
      input_features, norm1_w, norm1_b, normX, nullptr);
  ln_kernel<<<dim3(BATCH * SQLEN), dim3(256), 0, stream>>>(
      query_feature, norm2_w, norm2_b, normFLb, normFLf);
  transpose_bf16_kernel<<<dim3(EMBED / 32, SKPAD / 32, BATCH), dim3(256), 0, stream>>>(
      normX, normXT, SKLEN, SKPAD);
  flash_attn_kernel<<<dim3((SQLEN + 15) / 16, BATCH), dim3(128), 0, stream>>>(
      normFLb, normX, normXT, att, SQLEN, SKLEN, SKPAD);
  mlp_kernel<<<dim3(BATCH * SQLEN), dim3(256), 0, stream>>>(
      att, fc1_w, fc1_b, fc2_w, fc2_b, cross);
  ln_kernel<<<dim3(BATCH * SQLEN), dim3(256), 0, stream>>>(
      cross, norm3_w, norm3_b, ncrossb, nullptr);
  transpose_bf16_kernel<<<dim3(EMBED / 32, SQPAD / 32, BATCH), dim3(256), 0, stream>>>(
      ncrossb, ncrossT, SQLEN, SQPAD);
  flash_attn_kernel<<<dim3((SQLEN + 15) / 16, BATCH), dim3(128), 0, stream>>>(
      ncrossb, ncrossb, ncrossT, selfatt, SQLEN, SQLEN, SQPAD);
  const long long n = (long long)nQ;
  final_kernel<<<dim3((unsigned)((n + 255) / 256)), dim3(256), 0, stream>>>(
      normFLf, cross, selfatt, gamma1, gamma2, out, n);
}
